// MultiHeadAttention_85959475462224
// MI455X (gfx1250) — matrix operand order repaired and checked by reading, not compiled, not run on hardware
//
#include <hip/hip_runtime.h>
#include <hip/hip_bf16.h>

// ---------------------------------------------------------------------------
// MHA forward for MI455X (gfx1250), wave32, WMMA f32_16x16x32_f16.
// B=4, S=2048, D=1024, H=16, HD=64. All GEMM-like work on v_wmma.
// ---------------------------------------------------------------------------

typedef _Float16 h16 __attribute__((ext_vector_type(16)));
typedef _Float16 h8  __attribute__((ext_vector_type(8)));
typedef float    f8  __attribute__((ext_vector_type(8)));

constexpr int Bc = 4, Sc = 2048, Dc = 1024, Hc = 16, HDc = 64;
constexpr int Mtot = Bc * Sc;            // 8192 tokens

__device__ __forceinline__ f8 wmma_f16(h16 a, h16 b, f8 c) {
  return __builtin_amdgcn_wmma_f32_16x16x32_f16(false, a, false, b, (short)0, c,
                                                false, false);
}

// A fragment (16x32, f16 source, row-major): lane<16 -> K {0..7,16..23},
// lane>=16 -> K {8..15,24..31}; row = r0 + (lane&15).
__device__ __forceinline__ h16 ld_a16(const _Float16* base, int ld, int r0, int k0) {
  const int lane = threadIdx.x & 31;
  const _Float16* p = base + (size_t)(r0 + (lane & 15)) * ld + k0 + ((lane >> 4) << 3);
  union { h16 v; h8 h[2]; } u;
  u.h[0] = *(const h8*)(p);
  u.h[1] = *(const h8*)(p + 16);
  return u.v;
}

// A fragment from f32 source (convert on load).
__device__ __forceinline__ h16 ld_a32(const float* base, int ld, int r0, int k0) {
  const int lane = threadIdx.x & 31;
  const float* p = base + (size_t)(r0 + (lane & 15)) * ld + k0 + ((lane >> 4) << 3);
  h16 r;
#pragma unroll
  for (int i = 0; i < 8; ++i) {
    r[i]     = (_Float16)p[i];
    r[8 + i] = (_Float16)p[16 + i];
  }
  return r;
}

// B fragment (32x16): lane holds column (c0 + lane&15); lane<16 -> K {0..7,16..23},
// lane>=16 -> K {8..15,24..31}. Source stored column-major wrt B,
// i.e. row (c0+lane) of the given matrix, contiguous over K.
__device__ __forceinline__ h16 ld_b16(const _Float16* base, int ld, int c0, int k0) {
  const int lane = threadIdx.x & 31;
  const _Float16* p = base + (size_t)(c0 + (lane & 15)) * ld + k0 + ((lane >> 4) << 3);
  union { h16 v; h8 h[2]; } u;
  u.h[0] = *(const h8*)(p);
  u.h[1] = *(const h8*)(p + 16);
  return u.v;
}

// --------------------------- weight f32 -> f16 ------------------------------
__global__ __launch_bounds__(256) void cvt_f16_kernel(const float* __restrict__ src,
                                                      _Float16* __restrict__ dst, int n) {
  int i = blockIdx.x * 256 + threadIdx.x;
  if (i < n) dst[i] = (_Float16)src[i];
}

// --------------------------- projection GEMM --------------------------------
// out = (x @ W^T + b) * scale. Block = 8 waves, tile 32(M) x 256(N).
// vmode 0: head-major out[bh][s][64]; vmode 1: transposed out[bh][64][S].
__global__ __launch_bounds__(256) void proj_kernel(const float* __restrict__ x,
                                                   const _Float16* __restrict__ W,
                                                   const float* __restrict__ bias,
                                                   _Float16* __restrict__ out,
                                                   float scale, int vmode) {
  const int lane = threadIdx.x & 31;
  const int w = threadIdx.x >> 5;
  const int m0 = blockIdx.x * 32 + (w & 1) * 16;
  const int n0 = blockIdx.y * 256 + (w >> 1) * 64;

  f8 acc[4] = {f8{}, f8{}, f8{}, f8{}};
  for (int k0 = 0; k0 < Dc; k0 += 32) {
    h16 a = ld_a32(x, Dc, m0, k0);
#pragma unroll
    for (int j = 0; j < 4; ++j) {
      h16 b = ld_b16(W, Dc, n0 + 16 * j, k0);
      acc[j] = wmma_f16(a, b, acc[j]);
    }
  }

  const int col = lane & 15, hi = lane >> 4;
  const int bb = m0 >> 11;             // batch (m0 multiple of 16, S=2048)
  if (vmode == 0) {
#pragma unroll
    for (int j = 0; j < 4; ++j) {
      const int n = n0 + 16 * j + col;
      const int h = n >> 6, dc = n & 63;
      const float bn = bias[n];
#pragma unroll
      for (int r = 0; r < 8; ++r) {
        const int m = m0 + r + 8 * hi;
        const int s = m & (Sc - 1);
        out[((size_t)(bb * Hc + h) * Sc + s) * HDc + dc] =
            (_Float16)((acc[j][r] + bn) * scale);
      }
    }
  } else {
    // transposed store: out[bh][dcol][S]; 8 consecutive tokens -> one b128
    const int s0 = (m0 & (Sc - 1)) + 8 * hi;
#pragma unroll
    for (int j = 0; j < 4; ++j) {
      const int n = n0 + 16 * j + col;
      const int h = n >> 6, dc = n & 63;
      const float bn = bias[n];
      h8 v;
#pragma unroll
      for (int r = 0; r < 8; ++r) v[r] = (_Float16)((acc[j][r] + bn) * scale);
      *(h8*)(out + ((size_t)(bb * Hc + h) * HDc + dc) * Sc + s0) = v;
    }
  }
}

// ------------------------------ attention -----------------------------------
// One wave = 16 query rows x full HD=64. Online softmax over S keys in blocks
// of 32. P bounced through LDS (C-layout write -> A-layout b128 read).
__global__ __launch_bounds__(256) void attn_kernel(const _Float16* __restrict__ Qh,
                                                   const _Float16* __restrict__ Kh,
                                                   const _Float16* __restrict__ Vt,
                                                   _Float16* __restrict__ ctx) {
  __shared__ __attribute__((aligned(16))) _Float16 pbuf[8][16][40]; // pad: 40 halves/row
  const int lane = threadIdx.x & 31;
  const int w = threadIdx.x >> 5;
  const int bh = blockIdx.y;
  const int q0 = (blockIdx.x * 8 + w) * 16;

  const _Float16* Q = Qh + (size_t)bh * Sc * HDc;
  const _Float16* K = Kh + (size_t)bh * Sc * HDc;
  const _Float16* V = Vt + (size_t)bh * HDc * Sc;   // [64][S]

  const h16 aq0 = ld_a16(Q, HDc, q0, 0);
  const h16 aq1 = ld_a16(Q, HDc, q0, 32);

  f8 acc[4] = {f8{}, f8{}, f8{}, f8{}};
  float mrow[8], lrow[8];
#pragma unroll
  for (int r = 0; r < 8; ++r) { mrow[r] = -3.0e38f; lrow[r] = 0.0f; }

  const int prow = lane & 15;
  const int pcb = (lane >> 4) << 3;

  for (int j = 0; j < Sc; j += 32) {
    __builtin_prefetch(K + (size_t)(j + 32) * HDc, 0, 1);
    // scores: two 16x16 tiles (keys j..j+15, j+16..j+31), contraction d=64
    f8 s0 = {}, s1 = {};
    s0 = wmma_f16(aq0, ld_b16(K, HDc, j, 0), s0);
    s0 = wmma_f16(aq1, ld_b16(K, HDc, j, 32), s0);
    s1 = wmma_f16(aq0, ld_b16(K, HDc, j + 16, 0), s1);
    s1 = wmma_f16(aq1, ld_b16(K, HDc, j + 16, 32), s1);

    // online softmax; row M = r + 8*(lane>=16), cols = lane&15 (+16 for s1)
#pragma unroll
    for (int r = 0; r < 8; ++r) {
      float mx = fmaxf(s0[r], s1[r]);
#pragma unroll
      for (int off = 1; off < 16; off <<= 1) mx = fmaxf(mx, __shfl_xor(mx, off, 32));
      const float mnew = fmaxf(mrow[r], mx);
      const float p0 = __expf(s0[r] - mnew);
      const float p1 = __expf(s1[r] - mnew);
      const float corr = __expf(mrow[r] - mnew);
      float ps = p0 + p1;
#pragma unroll
      for (int off = 1; off < 16; off <<= 1) ps += __shfl_xor(ps, off, 32);
      lrow[r] = lrow[r] * corr + ps;
      mrow[r] = mnew;
#pragma unroll
      for (int d = 0; d < 4; ++d) acc[d][r] *= corr;
      const int row = r + 8 * (lane >> 4);
      const int c = lane & 15;
      pbuf[w][row][c] = (_Float16)p0;
      pbuf[w][row][c + 16] = (_Float16)p1;
    }
    asm volatile("s_wait_dscnt 0x0" ::: "memory");   // intra-wave LDS RAW

    // read P back as 16x32 A fragment
    union { h16 v; h8 h[2]; } u;
    u.h[0] = *(const h8*)&pbuf[w][prow][pcb];
    u.h[1] = *(const h8*)&pbuf[w][prow][pcb + 16];
    const h16 pa = u.v;

    // ctx += P @ V  (B fragments contiguous thanks to transposed V)
#pragma unroll
    for (int d = 0; d < 4; ++d)
      acc[d] = wmma_f16(pa, ld_b16(V, Sc, d * 16, j), acc[d]);
  }

  // epilogue: divide by l, store ctx token-major f16 [b][s][h*64+d]
  const int bb = bh >> 4, h = bh & 15;
  const int c = lane & 15, hi = lane >> 4;
#pragma unroll
  for (int d = 0; d < 4; ++d)
#pragma unroll
    for (int r = 0; r < 8; ++r) {
      const int q = q0 + r + 8 * hi;
      ctx[(size_t)(bb * Sc + q) * Dc + h * HDc + d * 16 + c] =
          (_Float16)(acc[d][r] / lrow[r]);
    }
}

// --------------------------- output projection ------------------------------
__global__ __launch_bounds__(256) void oproj_kernel(const _Float16* __restrict__ ctx,
                                                    const _Float16* __restrict__ W,
                                                    const float* __restrict__ bias,
                                                    float* __restrict__ out) {
  const int lane = threadIdx.x & 31;
  const int w = threadIdx.x >> 5;
  const int m0 = blockIdx.x * 32 + (w & 1) * 16;
  const int n0 = blockIdx.y * 256 + (w >> 1) * 64;

  f8 acc[4] = {f8{}, f8{}, f8{}, f8{}};
  for (int k0 = 0; k0 < Dc; k0 += 32) {
    h16 a = ld_a16(ctx, Dc, m0, k0);
#pragma unroll
    for (int j = 0; j < 4; ++j)
      acc[j] = wmma_f16(a, ld_b16(W, Dc, n0 + 16 * j, k0), acc[j]);
  }
  const int col = lane & 15, hi = lane >> 4;
#pragma unroll
  for (int j = 0; j < 4; ++j) {
    const int n = n0 + 16 * j + col;
    const float bn = bias[n];
#pragma unroll
    for (int r = 0; r < 8; ++r)
      out[(size_t)(m0 + r + 8 * hi) * Dc + n] = acc[j][r] + bn;
  }
}

// ------------------------------- launcher -----------------------------------
extern "C" void kernel_launch(void* const* d_in, const int* in_sizes, int n_in,
                              void* d_out, int out_size, void* d_ws, size_t ws_size,
                              hipStream_t stream) {
  const float* q  = (const float*)d_in[0];
  const float* k  = (const float*)d_in[1];
  const float* v  = (const float*)d_in[2];
  const float* Wq = (const float*)d_in[3];
  const float* bq = (const float*)d_in[4];
  const float* Wk = (const float*)d_in[5];
  const float* bk = (const float*)d_in[6];
  const float* Wv = (const float*)d_in[7];
  const float* bv = (const float*)d_in[8];
  const float* Wo = (const float*)d_in[9];
  const float* bo = (const float*)d_in[10];

  _Float16* ws   = (_Float16*)d_ws;
  const size_t WN = (size_t)Dc * Dc;        // 1M halves per weight
  const size_t AN = (size_t)Mtot * Dc;      // 8M halves per activation
  _Float16* Wq16 = ws;
  _Float16* Wk16 = Wq16 + WN;
  _Float16* Wv16 = Wk16 + WN;
  _Float16* Wo16 = Wv16 + WN;
  _Float16* Qh   = Wo16 + WN;
  _Float16* Kh   = Qh + AN;
  _Float16* Vt   = Kh + AN;
  _Float16* Ctx  = Vt + AN;

  const int wn = Dc * Dc;
  cvt_f16_kernel<<<wn / 256, 256, 0, stream>>>(Wq, Wq16, wn);
  cvt_f16_kernel<<<wn / 256, 256, 0, stream>>>(Wk, Wk16, wn);
  cvt_f16_kernel<<<wn / 256, 256, 0, stream>>>(Wv, Wv16, wn);
  cvt_f16_kernel<<<wn / 256, 256, 0, stream>>>(Wo, Wo16, wn);

  const dim3 gGemm(Mtot / 32, Dc / 256);
  const float SCALE = 0.125f;               // 1/sqrt(64)
  proj_kernel<<<gGemm, 256, 0, stream>>>(q, Wq16, bq, Qh, SCALE, 0);
  proj_kernel<<<gGemm, 256, 0, stream>>>(k, Wk16, bk, Kh, 1.0f, 0);
  proj_kernel<<<gGemm, 256, 0, stream>>>(v, Wv16, bv, Vt, 1.0f, 1);

  attn_kernel<<<dim3(Sc / 16 / 8, Bc * Hc), 256, 0, stream>>>(Qh, Kh, Vt, Ctx);

  oproj_kernel<<<gGemm, 256, 0, stream>>>(Ctx, Wo16, bo, (float*)d_out);
}
